// ViTCCSegmentation_63625645523319
// MI455X (gfx1250) — matrix operand order repaired and checked by reading, not compiled, not run on hardware
//
#include <hip/hip_runtime.h>
#include <cstdint>
#include <cstddef>

// ---------------------------------------------------------------------------
// Types / WMMA helper (CDNA5 gfx1250, wave32)
// ---------------------------------------------------------------------------
typedef __attribute__((ext_vector_type(16))) _Float16 v16h;
typedef __attribute__((ext_vector_type(8)))  float    v8f;

union V16 { v16h v; _Float16 h[16]; uint32_t u[8]; };

static __device__ inline v8f wmma_f16(v16h a, v16h b, v8f c) {
  // D = A(16x32 f16) * B(32x16 f16) + C(16x16 f32)
  return __builtin_amdgcn_wmma_f32_16x16x32_f16(false, a, false, b, (short)0, c,
                                                false, false);
}

#define F_RELU 1
#define F_RES  2
#define F_POS  4

// ---------------------------------------------------------------------------
// Generic WMMA GEMM:  C[M,N] = A[M,K](f16,row-major) * W[N,K](f16,row-major)^T
// one wave per 16x16 tile, 8 waves per block. K % 32 == 0, M,N % 16 == 0.
// Epilogue: +bias*scale, +pos_embed, ReLU, residual add (f32), f16 store.
// ---------------------------------------------------------------------------
__global__ __launch_bounds__(256) void k_gemm(
    const _Float16* __restrict__ A, const _Float16* __restrict__ W,
    const float* __restrict__ bias, float bias_scale,
    const float* __restrict__ pos,
    float* __restrict__ outF, _Float16* __restrict__ outH,
    int M, int N, int K, int flags)
{
  int wave = threadIdx.x >> 5;
  int lane = threadIdx.x & 31;
  int ntN  = N >> 4;
  int tile = blockIdx.x * 8 + wave;
  if (tile >= (M >> 4) * ntN) return;           // wave-uniform exit: EXEC stays all-1
  int tm = tile / ntN, tn = tile % ntN;
  int lo = lane >> 4, c16 = lane & 15;

  const _Float16* Arow = A + (size_t)(tm * 16 + c16) * K;
  const _Float16* Wrow = W + (size_t)(tn * 16 + c16) * K + lo * 16;

  v8f acc = {};
  for (int k0 = 0; k0 < K; k0 += 32) {
    V16 a, b;
    // A 16x32 f16 fragment: lane = row, vgpr v holds K pair
    //   K = (lane>=16?16:0) + v*2 + {0,1}
#pragma unroll
    for (int v = 0; v < 8; ++v) {
      int kk = k0 + lo * 16 + (v << 1);
      a.u[v] = *(const uint32_t*)(Arow + kk);
    }
    // B 32x16 f16 fragment: lane%16 = col, K = (lane/16)*16 + e  -> 32B contiguous
    b.v = *(const v16h*)(Wrow + k0);
    acc = wmma_f16(a.v, b.v, acc);
  }

  int cb = tn * 16 + c16;
  float bv = bias ? bias_scale * bias[cb] : 0.f;
#pragma unroll
  for (int r = 0; r < 8; ++r) {
    int rr = tm * 16 + r + lo * 8;              // C/D layout: vgpr r -> M = r (+8 hi)
    float val = acc[r] + bv;
    if (flags & F_POS)  val += pos[(size_t)(rr & 255) * N + cb];
    if (flags & F_RELU) val = fmaxf(val, 0.f);
    size_t o = (size_t)rr * N + cb;
    if (flags & F_RES)      outF[o] += val;
    else if (outF)          outF[o]  = val;
    if (outH)               outH[o]  = (_Float16)val;
  }
}

// ---------------------------------------------------------------------------
// Cross-column attention: one wave per (dir, b, seq), loops 12 heads.
// S=16, dh=64.  QK^T: 2 WMMAs (K=64).  P*V: 4 WMMAs (K padded 16->32).
// Writes pre-projection head outputs (disjoint per dir) + attn column stats.
// ---------------------------------------------------------------------------
__global__ __launch_bounds__(32) void k_attn(
    const float* __restrict__ qkv,              // [512, 2304] (q|k|v)
    float* __restrict__ Orow, float* __restrict__ Ocol,   // [512, 768]
    float* __restrict__ statrow, float* __restrict__ statcol) // [B*16, 16]
{
  __shared__ float    sc[16][16];
  __shared__ _Float16 psh[16][16];
  __shared__ _Float16 vsh[16][64];
  __shared__ float    statL[16];

  int lane = threadIdx.x;
  int blk  = blockIdx.x;                        // 2*2*16 = 64
  int dir  = blk >> 5;                          // 0 = row, 1 = col
  int rem  = blk & 31;
  int b    = rem >> 4;
  int m    = rem & 15;                          // sequence id
  float* O     = dir ? Ocol   : Orow;
  float* statg = dir ? statcol : statrow;
  int lo = lane >> 4, c16 = lane & 15;
  int bb = b * 256;
  auto tok = [&](int s) -> int { return dir == 0 ? bb + s * 16 + m : bb + m * 16 + s; };

  if (lane < 16) statL[lane] = 0.f;

  for (int hh = 0; hh < 12; ++hh) {
    // ---- Q fragments (A-layout), K fragments (B-layout), d = head dim ----
    V16 qa0, qa1, kb0, kb1;
    const float* qb = qkv + (size_t)tok(c16) * 2304 + hh * 64;
    const float* kb = qkv + (size_t)tok(c16) * 2304 + 768 + hh * 64 + lo * 16;
#pragma unroll
    for (int e = 0; e < 16; ++e) {
      int v  = e >> 1;
      int kk = lo * 16 + e;
      qa0.h[e] = (_Float16)qb[kk];
      qa1.h[e] = (_Float16)qb[32 + kk];
      kb0.h[e] = (_Float16)kb[e];
      kb1.h[e] = (_Float16)kb[32 + e];
    }
    v8f s8 = {};
    s8 = wmma_f16(qa0.v, kb0.v, s8);
    s8 = wmma_f16(qa1.v, kb1.v, s8);
#pragma unroll
    for (int r = 0; r < 8; ++r) sc[r + lo * 8][c16] = s8[r];
    __syncthreads();

    // ---- softmax per row (lanes 0-15 each own one query row) ----
    if (lane < 16) {
      float mx = -1e30f;
#pragma unroll
      for (int t = 0; t < 16; ++t) mx = fmaxf(mx, sc[lane][t]);
      float ex[16], sum = 0.f;
#pragma unroll
      for (int t = 0; t < 16; ++t) { ex[t] = __expf((sc[lane][t] - mx) * 0.125f); sum += ex[t]; }
      float inv = 1.f / sum;
#pragma unroll
      for (int t = 0; t < 16; ++t) psh[lane][t] = (_Float16)(ex[t] * inv);
    }
    // ---- stage V tile in LDS ----
#pragma unroll
    for (int i = 0; i < 32; ++i) {
      int idx = lane + i * 32;
      int t = idx >> 6, d = idx & 63;
      vsh[t][d] = (_Float16)qkv[(size_t)tok(t) * 2304 + 1536 + hh * 64 + d];
    }
    __syncthreads();

    // ---- attn column stats: mean over queries & heads ----
    if (lane < 16) {
      float cs = 0.f;
#pragma unroll
      for (int r = 0; r < 16; ++r) cs += (float)psh[r][lane];
      statL[lane] += cs * (1.f / 192.f);        // /16 queries /12 heads
    }

    // ---- P (16x16, zero-padded to K=32) x V (16x64) ----
    V16 pa;
#pragma unroll
    for (int e = 0; e < 16; ++e) {
      int v  = e >> 1;
      int kk = lo * 16 + e;
      pa.h[e] = (kk < 16) ? psh[c16][kk] : (_Float16)0.f;
    }
#pragma unroll
    for (int c = 0; c < 4; ++c) {
      V16 vb;
#pragma unroll
      for (int e = 0; e < 16; ++e) {
        int t = lo * 16 + e;                    // K dim = key position
        vb.h[e] = (t < 16) ? vsh[t][c * 16 + c16] : (_Float16)0.f;
      }
      v8f o8 = {};
      o8 = wmma_f16(pa.v, vb.v, o8);
#pragma unroll
      for (int r = 0; r < 8; ++r)
        O[(size_t)tok(r + lo * 8) * 768 + hh * 64 + c * 16 + c16] = o8[r];
    }
    __syncthreads();
  }
  if (lane < 16) statg[(b * 16 + m) * 16 + lane] = statL[lane];
}

// ---------------------------------------------------------------------------
// LayerNorm over D=768, one block (256 threads) per token, f16 output.
// ---------------------------------------------------------------------------
__global__ __launch_bounds__(256) void k_ln(
    const float* __restrict__ h, const float* __restrict__ g,
    const float* __restrict__ b, _Float16* __restrict__ out)
{
  __shared__ float red[256];
  __shared__ float s_mean, s_rstd;
  int t = blockIdx.x, tid = threadIdx.x;
  const float* hr = h + (size_t)t * 768;
  float x0 = hr[tid], x1 = hr[tid + 256], x2 = hr[tid + 512];
  red[tid] = x0 + x1 + x2; __syncthreads();
  for (int o = 128; o > 0; o >>= 1) { if (tid < o) red[tid] += red[tid + o]; __syncthreads(); }
  if (tid == 0) s_mean = red[0] * (1.f / 768.f);
  __syncthreads();
  float mean = s_mean;
  float d0 = x0 - mean, d1 = x1 - mean, d2 = x2 - mean;
  red[tid] = d0 * d0 + d1 * d1 + d2 * d2; __syncthreads();
  for (int o = 128; o > 0; o >>= 1) { if (tid < o) red[tid] += red[tid + o]; __syncthreads(); }
  if (tid == 0) s_rstd = rsqrtf(red[0] * (1.f / 768.f) + 1e-5f);
  __syncthreads();
  float rs = s_rstd;
  _Float16* orow = out + (size_t)t * 768;
  orow[tid]       = (_Float16)(d0 * rs * g[tid]       + b[tid]);
  orow[tid + 256] = (_Float16)(d1 * rs * g[tid + 256] + b[tid + 256]);
  orow[tid + 512] = (_Float16)(d2 * rs * g[tid + 512] + b[tid + 512]);
}

// ---------------------------------------------------------------------------
// Small utility kernels
// ---------------------------------------------------------------------------
__global__ void k_patchify(const float* __restrict__ x, _Float16* __restrict__ A)
{ // A[t=b*256+gy*16+gx][c*256+py*16+px] = x[b][c][gy*16+py][gx*16+px]
  int idx = blockIdx.x * 256 + threadIdx.x;
  if (idx >= 512 * 768) return;
  int t = idx / 768, col = idx - t * 768;
  int b = t >> 8, n = t & 255, gy = n >> 4, gx = n & 15;
  int c = col >> 8, py = (col >> 4) & 15, px = col & 15;
  A[idx] = (_Float16)x[(((size_t)(b * 3 + c) * 256) + gy * 16 + py) * 256 + gx * 16 + px];
}

__global__ void k_cvt(const float* __restrict__ s, _Float16* __restrict__ d, int n)
{ int i = blockIdx.x * 256 + threadIdx.x; if (i < n) d[i] = (_Float16)s[i]; }

__global__ void k_cvt_conv(const float* __restrict__ w, _Float16* __restrict__ d)
{ // d[n=co*9+j*3+i][k=ci] = conv1_w[co][ci][j][i]
  int idx = blockIdx.x * 256 + threadIdx.x;
  if (idx >= 2304 * 768) return;
  int n = idx / 768, k = idx - n * 768;
  int co = n / 9, ji = n - co * 9;
  d[idx] = (_Float16)w[((size_t)co * 768 + k) * 9 + ji];
}

__global__ void k_add2f16(const float* __restrict__ a, const float* __restrict__ b,
                          _Float16* __restrict__ o, int n)
{ int i = blockIdx.x * 256 + threadIdx.x; if (i < n) o[i] = (_Float16)(a[i] + b[i]); }

__global__ void k_zero(float* __restrict__ p, int n)
{ int i = blockIdx.x * 256 + threadIdx.x; if (i < n) p[i] = 0.f; }

__global__ void k_edge(const float* __restrict__ sr, const float* __restrict__ sc,
                       float* __restrict__ divsum)
{ // divsum[b*256 + s1*16 + s2] += |rowstat - colstat|
  int i = blockIdx.x * 256 + threadIdx.x;
  if (i >= 512) return;
  int b = i >> 8, r = i & 255, s1 = r >> 4, s2 = r & 15;
  divsum[i] += fabsf(sr[(b * 16 + s1) * 16 + s2] - sc[(b * 16 + s1) * 16 + s2]);
}

__global__ void k_finalfeat(const float* __restrict__ h, const float* __restrict__ divsum,
                            const float* __restrict__ ew, const float* __restrict__ eb,
                            _Float16* __restrict__ A)
{ // A = f16( h + divsum/L * edge_w + edge_b )
  int idx = blockIdx.x * 256 + threadIdx.x;
  if (idx >= 512 * 768) return;
  int t = idx / 768, d = idx - t * 768;
  A[idx] = (_Float16)(h[idx] + divsum[t] * 0.125f * ew[d] + eb[d]);
}

// ---------------------------------------------------------------------------
// Fused bilinear-upsample + 3x3 conv taps + ReLU + 1x1 conv2.
// Z[b, gy, gx, co*9+j*3+i] already holds W(3x3)∘channel-contracted features.
// Each thread = one output pixel: 9 taps x 4 bilinear MACs per out-channel.
// ---------------------------------------------------------------------------
__global__ __launch_bounds__(256) void k_upconv(
    const float* __restrict__ Z, const float* __restrict__ c1b,
    const float* __restrict__ c2w, const float* __restrict__ c2b,
    float* __restrict__ out)
{
  int idx = blockIdx.x * 256 + threadIdx.x;     // 2*256*256
  if (idx >= 131072) return;
  int X = idx & 255, Y = (idx >> 8) & 255, b = idx >> 16;

  float wxa[3], wxb[3], wya[3], wyb[3];
  int   ixa[3], ixb[3], iya[3], iyb[3];
  bool  vx[3], vy[3];
#pragma unroll
  for (int i = 0; i < 3; ++i) {
    int Xp = X + i - 1;
    vx[i] = (Xp >= 0) && (Xp < 256);
    float sx = (Xp + 0.5f) * 0.0625f - 0.5f;    // half-pixel, scale 16
    float fl = floorf(sx); float f = sx - fl;
    int g0 = (int)fl, g1 = g0 + 1;
    g0 = g0 < 0 ? 0 : (g0 > 15 ? 15 : g0);
    g1 = g1 < 0 ? 0 : (g1 > 15 ? 15 : g1);
    wxa[i] = 1.f - f; wxb[i] = f; ixa[i] = g0; ixb[i] = g1;

    int Yp = Y + i - 1;
    vy[i] = (Yp >= 0) && (Yp < 256);
    float sy = (Yp + 0.5f) * 0.0625f - 0.5f;
    fl = floorf(sy); f = sy - fl;
    g0 = (int)fl; g1 = g0 + 1;
    g0 = g0 < 0 ? 0 : (g0 > 15 ? 15 : g0);
    g1 = g1 < 0 ? 0 : (g1 > 15 ? 15 : g1);
    wya[i] = 1.f - f; wyb[i] = f; iya[i] = g0; iyb[i] = g1;
  }

  const float* Zb = Z + (size_t)b * 256 * 2304;
  float accO = 0.f;
  for (int co = 0; co < 256; ++co) {
    float a = c1b[co];
    int nb = co * 9;
#pragma unroll
    for (int j = 0; j < 3; ++j) {
      if (!vy[j]) continue;
      const float* r0 = Zb + (size_t)(iya[j] * 16) * 2304;
      const float* r1 = Zb + (size_t)(iyb[j] * 16) * 2304;
#pragma unroll
      for (int i = 0; i < 3; ++i) {
        if (!vx[i]) continue;
        int zo = nb + j * 3 + i;
        float t0 = wxa[i] * r0[(size_t)ixa[i] * 2304 + zo] + wxb[i] * r0[(size_t)ixb[i] * 2304 + zo];
        float t1 = wxa[i] * r1[(size_t)ixa[i] * 2304 + zo] + wxb[i] * r1[(size_t)ixb[i] * 2304 + zo];
        a += wya[j] * t0 + wyb[j] * t1;
      }
    }
    a = fmaxf(a, 0.f);                          // ReLU after conv1
    accO += a * c2w[co];                        // fused 1x1 conv2
  }
  out[idx] = accO + c2b[0];
}

// ---------------------------------------------------------------------------
// Host orchestration
// ---------------------------------------------------------------------------
extern "C" void kernel_launch(void* const* d_in, const int* in_sizes, int n_in,
                              void* d_out, int out_size, void* d_ws, size_t ws_size,
                              hipStream_t stream) {
  (void)in_sizes; (void)n_in; (void)out_size; (void)ws_size;
  const float* x        = (const float*)d_in[0];
  const float* patch_w  = (const float*)d_in[1];
  const float* patch_b  = (const float*)d_in[2];
  const float* pos_emb  = (const float*)d_in[3];
  const float* in_pw    = (const float*)d_in[4];
  const float* in_pb    = (const float*)d_in[5];
  const float* out_w    = (const float*)d_in[6];
  const float* out_b    = (const float*)d_in[7];
  const float* ln1_g    = (const float*)d_in[8];
  const float* ln1_b    = (const float*)d_in[9];
  const float* ln2_g    = (const float*)d_in[10];
  const float* ln2_b    = (const float*)d_in[11];
  const float* ffn_w1   = (const float*)d_in[12];
  const float* ffn_b1   = (const float*)d_in[13];
  const float* ffn_w2   = (const float*)d_in[14];
  const float* ffn_b2   = (const float*)d_in[15];
  const float* edge_w   = (const float*)d_in[16];
  const float* edge_b   = (const float*)d_in[17];
  const float* conv1_w  = (const float*)d_in[18];
  const float* conv1_b  = (const float*)d_in[19];
  const float* conv2_w  = (const float*)d_in[20];
  const float* conv2_b  = (const float*)d_in[21];
  float* outp = (float*)d_out;

  // --- workspace partition (≈25 MB), 256B aligned ---
  char* base = (char*)d_ws; size_t off = 0;
  auto alloc = [&](size_t bytes) -> void* {
    void* r = base + off; off += (bytes + 255) & ~(size_t)255; return r;
  };
  _Float16* wf16   = (_Float16*)alloc((size_t)2304 * 768 * 2);  // staged f16 weights (reused)
  _Float16* patchA = (_Float16*)alloc((size_t)512 * 768 * 2);
  _Float16* lnA    = (_Float16*)alloc((size_t)512 * 768 * 2);
  _Float16* ffnA   = (_Float16*)alloc((size_t)512 * 2048 * 2);
  _Float16* OcA    = (_Float16*)alloc((size_t)512 * 768 * 2);
  _Float16* featA  = (_Float16*)alloc((size_t)512 * 768 * 2);
  float* hbuf   = (float*)alloc((size_t)512 * 768 * 4);
  float* qkv    = (float*)alloc((size_t)512 * 2304 * 4);
  float* Orow   = (float*)alloc((size_t)512 * 768 * 4);
  float* Ocol   = (float*)alloc((size_t)512 * 768 * 4);
  float* Zbuf   = (float*)alloc((size_t)512 * 2304 * 4);
  float* strow  = (float*)alloc((size_t)512 * 4);
  float* stcol  = (float*)alloc((size_t)512 * 4);
  float* divsum = (float*)alloc((size_t)512 * 4);

  auto gemm = [&](const _Float16* A, const _Float16* W, const float* bias, float bs,
                  const float* pos, float* oF, _Float16* oH, int M, int N, int K, int fl) {
    int ntiles = (M / 16) * (N / 16);
    k_gemm<<<(ntiles + 7) / 8, 256, 0, stream>>>(A, W, bias, bs, pos, oF, oH, M, N, K, fl);
  };
  auto cvt = [&](const float* s, _Float16* d, int n) {
    k_cvt<<<(n + 255) / 256, 256, 0, stream>>>(s, d, n);
  };

  // --- patch embedding: h = patches @ patch_w^T + patch_b + pos_embed ---
  k_patchify<<<1536, 256, 0, stream>>>(x, patchA);
  cvt(patch_w, wf16, 768 * 768);
  gemm(patchA, wf16, patch_b, 1.f, pos_emb, hbuf, nullptr, 512, 768, 768, F_POS);
  k_zero<<<2, 256, 0, stream>>>(divsum, 512);

  // --- 8 transformer blocks with cross-column attention ---
  for (int i = 0; i < 8; ++i) {
    k_ln<<<512, 256, 0, stream>>>(hbuf, ln1_g + i * 768, ln1_b + i * 768, lnA);
    cvt(in_pw + (size_t)i * 2304 * 768, wf16, 2304 * 768);
    gemm(lnA, wf16, in_pb + i * 2304, 1.f, nullptr, qkv, nullptr, 512, 2304, 768, 0);
    k_attn<<<64, 32, 0, stream>>>(qkv, Orow, Ocol, strow, stcol);
    k_add2f16<<<1536, 256, 0, stream>>>(Orow, Ocol, OcA, 512 * 768);
    cvt(out_w + (size_t)i * 768 * 768, wf16, 768 * 768);
    // row_out + col_out both go through the same out-proj: bias counted twice
    gemm(OcA, wf16, out_b + i * 768, 2.f, nullptr, hbuf, nullptr, 512, 768, 768, F_RES);
    k_edge<<<2, 256, 0, stream>>>(strow, stcol, divsum);
    k_ln<<<512, 256, 0, stream>>>(hbuf, ln2_g + i * 768, ln2_b + i * 768, lnA);
    cvt(ffn_w1 + (size_t)i * 2048 * 768, wf16, 2048 * 768);
    gemm(lnA, wf16, ffn_b1 + i * 2048, 1.f, nullptr, nullptr, ffnA, 512, 2048, 768, F_RELU);
    cvt(ffn_w2 + (size_t)i * 768 * 2048, wf16, 768 * 2048);
    gemm(ffnA, wf16, ffn_b2 + i * 768, 1.f, nullptr, hbuf, nullptr, 512, 768, 2048, F_RES);
  }

  // --- edge tokens folded in; conv1 channel-contraction at grid resolution ---
  k_finalfeat<<<1536, 256, 0, stream>>>(hbuf, divsum, edge_w, edge_b, featA);
  k_cvt_conv<<<(2304 * 768 + 255) / 256, 256, 0, stream>>>(conv1_w, wf16);
  gemm(featA, wf16, nullptr, 0.f, nullptr, Zbuf, nullptr, 512, 2304, 768, 0);

  // --- fused upsample + conv1 taps + ReLU + conv2 ---
  k_upconv<<<512, 256, 0, stream>>>(Zbuf, conv1_b, conv2_w, conv2_b, outp);
}
